// MABSINK_Simplified_85744727098042
// MI455X (gfx1250) — matrix operand order repaired and checked by reading, not compiled, not run on hardware
//
#include <hip/hip_runtime.h>
#include <hip/hip_bf16.h>
#include <stdint.h>

// ---------------------------------------------------------------------------
// MABSINK (Sinkhorn attention block) for MI455X / gfx1250, wave32 + WMMA bf16
// + TDM (tensor_load_to_lds) staging of the per-head X tile.
// B=8, n=1024, dmodel=512, H=8, dhead=64.
// ---------------------------------------------------------------------------

typedef __attribute__((ext_vector_type(16))) __bf16   v16bf;
typedef __attribute__((ext_vector_type(8)))  __bf16   v8bf;
typedef __attribute__((ext_vector_type(8)))  float    v8f;
typedef __attribute__((ext_vector_type(4)))  uint32_t u32x4;
typedef __attribute__((ext_vector_type(8)))  int      i32x8;
typedef __attribute__((ext_vector_type(4)))  int      i32x4;

#if __has_include(<hip/amd_detail/amd_gfx1250_TDM.h>)
#define TDM_6ARG 1
#else
#define TDM_6ARG 0
#endif

#define WMMA_BF16(A, B, C) \
  __builtin_amdgcn_wmma_f32_16x16x32_bf16(false, (A), false, (B), (short)0, (C), false, false)

static __device__ __forceinline__ __bf16 f2bf(float f) {
  uint32_t u = __builtin_bit_cast(uint32_t, f);
  uint32_t r = (u + 0x7FFFu + ((u >> 16) & 1u)) >> 16;
  return __builtin_bit_cast(__bf16, (uint16_t)r);
}

static __device__ __forceinline__ v8bf ld8(const __bf16* p) {
  return *(const v8bf*)p;
}

static __device__ __forceinline__ v16bf cat16(v8bf lo, v8bf hi) {
  v16bf r;
#pragma unroll
  for (int i = 0; i < 8; ++i) { r[i] = lo[i]; r[8 + i] = hi[i]; }
  return r;
}

// ---------------------------------------------------------------------------
// fp32 -> bf16 convert
// ---------------------------------------------------------------------------
__global__ __launch_bounds__(256) void cvt_bf16_kernel(const float* __restrict__ in,
                                                       __bf16* __restrict__ out, int n) {
  int i = blockIdx.x * 256 + threadIdx.x;
  if (i < n) out[i] = f2bf(in[i]);
}

// ---------------------------------------------------------------------------
// Repack Qp into head-major, pre-scaled bf16 X:  Xg[hb][r][k] (contiguous
// 128KB per head-batch) so the attention kernel can TDM-load it in one shot.
// ---------------------------------------------------------------------------
__global__ __launch_bounds__(256) void repack_head_kernel(const float* __restrict__ Qp,
                                                          __bf16* __restrict__ Xg) {
  const int hb = blockIdx.x;   // hb = h*B + b
  const int h  = hb >> 3;
  const int b  = hb & 7;
  const int t  = threadIdx.x;
  const float invs = 0.21022410f;      // 512^-0.25
  for (int r0 = 0; r0 < 1024; r0 += 16) {
    const int r = r0 + (t >> 4);
    const int k = (t & 15) * 4;
    const float4 q = *(const float4*)(Qp + ((size_t)(b * 1024 + r)) * 512 + h * 64 + k);
    __bf16* o = Xg + ((size_t)hb * 1024 + r) * 64 + k;
    o[0] = f2bf(q.x * invs); o[1] = f2bf(q.y * invs);
    o[2] = f2bf(q.z * invs); o[3] = f2bf(q.w * invs);
  }
}

// ---------------------------------------------------------------------------
// C[M,512] = epilogue( Abf[M,512] @ Wbf[512,512]^T + bias ),  K = N = 512
// Block: 256 thr (8 waves).  Block tile 64 rows x 128 cols; wave tile 16x64.
// ---------------------------------------------------------------------------
__global__ __launch_bounds__(256) void gemm_bf16_kernel(
    const __bf16* __restrict__ A, const __bf16* __restrict__ W,
    const float* __restrict__ bias, const float* __restrict__ Res,
    float* __restrict__ C, int relu) {
  const int wave = threadIdx.x >> 5;
  const int lane = threadIdx.x & 31;
  const int l16  = lane & 15;
  const int hi   = lane >> 4;
  const int row0 = blockIdx.x * 64 + (wave & 3) * 16;
  const int col0 = blockIdx.y * 128 + (wave >> 2) * 64;

  v8f acc[4] = {};
  const __bf16* arow = A + (size_t)(row0 + l16) * 512;

  for (int kb = 0; kb < 512; kb += 32) {
    v16bf av = cat16(ld8(arow + kb + hi * 8), ld8(arow + kb + 16 + hi * 8));
#pragma unroll
    for (int cc = 0; cc < 4; ++cc) {
      const __bf16* wrow = W + (size_t)(col0 + cc * 16 + l16) * 512 + kb + hi * 16;
      v16bf bv = cat16(ld8(wrow - hi * 8), ld8(wrow - hi * 8 + 16));
      acc[cc] = WMMA_BF16(av, bv, acc[cc]);
    }
  }

#pragma unroll
  for (int cc = 0; cc < 4; ++cc) {
    const int col = col0 + cc * 16 + l16;
    const float bv = bias[col];
#pragma unroll
    for (int v = 0; v < 8; ++v) {
      const int row = row0 + v + 8 * hi;          // D layout: M = v + 8*hi
      float x = acc[cc][v] + bv;
      if (relu) x = fmaxf(x, 0.0f);
      if (Res)  x += Res[(size_t)row * 512 + col];
      C[(size_t)row * 512 + col] = x;
    }
  }
}

// ---------------------------------------------------------------------------
// Sinkhorn attention: one workgroup per (head, batch). 512 thr = 16 waves.
// LDS: Xb  [1024][72] bf16  <- filled by TDM tensor_load_to_lds (pad feature
//                              inserts the 8-element row padding in-flight)
//      Xt  [64][1032] bf16  (transposed copy built LDS->LDS)
//      Us  [1024] f32, Vs [1024] f32
//      Psc [16 waves][16][40] bf16 (D->A relayout scratch for P tiles)
// ---------------------------------------------------------------------------
#define XB_STRIDE 72
#define XT_STRIDE 1032
#define P_STRIDE  40
#define ATTN_SMEM ((1024 * XB_STRIDE + 64 * XT_STRIDE) * 2 + 2 * 1024 * 4 + 16 * 16 * P_STRIDE * 2)

__global__ __launch_bounds__(512) void attn_sinkhorn_kernel(
    const float* __restrict__ Qp, const __bf16* __restrict__ Xg,
    float* __restrict__ Opre) {
  extern __shared__ char smem[];
  __bf16* Xb = (__bf16*)smem;                       // 1024 x 72, offset 0 in LDS
  __bf16* Xt = Xb + 1024 * XB_STRIDE;               // 64 x 1032
  float*  Us = (float*)(Xt + 64 * XT_STRIDE);       // 1024
  float*  Vs = Us + 1024;                           // 1024
  __bf16* Psc = (__bf16*)(Vs + 1024);               // 16 x (16 x 40)

  const int hb = blockIdx.x;        // head-major stacking: hb = h*B + b
  const int h  = hb >> 3;
  const int b  = hb & 7;
  const int tid  = threadIdx.x;
  const int wave = tid >> 5;
  const int lane = tid & 31;
  const int l16  = lane & 15;
  const int hi   = lane >> 4;

  const float sc    = 4.75682846f;      // 512^+0.25
  const float logmu = __logf(1.0f / 1024.0f + 1e-8f);
  const float logn  = __logf(1024.0f);

  // ---- TDM: DMA this head's 1024x64 bf16 tile into Xb with row padding ----
  // pad_interval code 4 = 32 DWORDs (one 128B row), pad_amount code 3 = 4
  // DWORDs (16B) -> LDS row stride 144B = 72 bf16.
  if (wave == 0) {
    const uint64_t ga = (uint64_t)(uintptr_t)Xg + (uint64_t)hb * (1024u * 64u * 2u);
    u32x4 g0;
    g0[0] = 1u;                                   // count=1 (valid user D#)
    g0[1] = 0u;                                   // lds_addr = 0 (dyn LDS base)
    g0[2] = (uint32_t)ga;                         // global_addr[31:0]
    g0[3] = (uint32_t)((ga >> 32) & 0x01FFFFFFu)  // global_addr[56:32]
            | (2u << 30);                         // type = 2 ("image")
    i32x8 g1;
    g1[0] = (int)((1u << 16)      // data_size = 2 bytes
                  | (1u << 20)    // pad_enable
                  | (4u << 22)    // pad_interval: 32 DWORDs
                  | (3u << 25));  // pad_amount:   4 DWORDs
    g1[1] = (int)(64u << 16);     // tensor_dim0 = 64 (bits 79:48)
    g1[2] = (int)(1024u << 16);   // tensor_dim1 = 1024 (bits 111:80)
    g1[3] = (int)(64u << 16);     // tile_dim0 = 64 (bits 127:112)
    g1[4] = (int)1024u;           // tile_dim1 = 1024, tile_dim2 = 0
    g1[5] = (int)64u;             // tensor_dim0_stride = 64
    g1[6] = 0;                    // dim0_stride hi / dim1_stride lo
    g1[7] = 0;
    const i32x4 z4 = {0, 0, 0, 0};
#if TDM_6ARG
    const i32x8 z8 = {0, 0, 0, 0, 0, 0, 0, 0};
    __builtin_amdgcn_tensor_load_to_lds(g0, g1, z4, z4, z8, 0);
#else
    __builtin_amdgcn_tensor_load_to_lds(g0, g1, z4, z4, 0);
#endif
    __builtin_amdgcn_s_wait_tensorcnt(0);
  }
  __syncthreads();

  // ---- build transposed copy Xt[k][r] from Xb (LDS -> LDS) ----
  for (int i = tid; i < 1024 * 4; i += 512) {
    const int r = i >> 2;
    const int kc = (i & 3) * 16;
    const v8bf v0 = ld8(Xb + r * XB_STRIDE + kc);
    const v8bf v1 = ld8(Xb + r * XB_STRIDE + kc + 8);
#pragma unroll
    for (int e = 0; e < 8; ++e) {
      Xt[(kc + e) * XT_STRIDE + r]     = v0[e];
      Xt[(kc + 8 + e) * XT_STRIDE + r] = v1[e];
    }
  }
  __syncthreads();

  // ======== Pass A: u = log_mu - logsumexp_j(S_ij) ========
  for (int t = wave; t < 64; t += 16) {
    const int R = t * 16;
    const __bf16* arow = Xb + (R + l16) * XB_STRIDE;
    const v16bf a0 = cat16(ld8(arow + hi * 8),      ld8(arow + 16 + hi * 8));
    const v16bf a1 = cat16(ld8(arow + 32 + hi * 8), ld8(arow + 48 + hi * 8));
    float rm[8], rs[8];
#pragma unroll
    for (int v = 0; v < 8; ++v) { rm[v] = -1e30f; rs[v] = 0.0f; }

    for (int j = 0; j < 64; ++j) {
      const __bf16* brow = Xb + (j * 16 + l16) * XB_STRIDE;
      const v16bf b0 = cat16(ld8(brow + hi * 8),      ld8(brow + 16 + hi * 8));
      const v16bf b1 = cat16(ld8(brow + 32 + hi * 8), ld8(brow + 48 + hi * 8));
      v8f c = {};
      c = WMMA_BF16(a0, b0, c);
      c = WMMA_BF16(a1, b1, c);
#pragma unroll
      for (int v = 0; v < 8; ++v) {
        const float x  = c[v];
        const float nm = fmaxf(rm[v], x);
        rs[v] = rs[v] * __expf(rm[v] - nm) + __expf(x - nm);
        rm[v] = nm;
      }
    }
#pragma unroll
    for (int v = 0; v < 8; ++v) {
      float m = rm[v], s = rs[v];
#pragma unroll
      for (int mk = 1; mk < 16; mk <<= 1) {
        const float om = __shfl_xor(m, mk, 32);
        const float os = __shfl_xor(s, mk, 32);
        const float nm = fmaxf(m, om);
        s = s * __expf(m - nm) + os * __expf(om - nm);
        m = nm;
      }
      if (l16 == 0) Us[R + v + 8 * hi] = logmu - (m + __logf(s));
    }
  }
  __syncthreads();

  // ======== Pass B: v_j = log_mu - logsumexp_i(S_ij + u_i)  (S symmetric) ====
  for (int t = wave; t < 64; t += 16) {
    const int R = t * 16;
    const __bf16* arow = Xb + (R + l16) * XB_STRIDE;
    const v16bf a0 = cat16(ld8(arow + hi * 8),      ld8(arow + 16 + hi * 8));
    const v16bf a1 = cat16(ld8(arow + 32 + hi * 8), ld8(arow + 48 + hi * 8));
    float rm[8], rs[8];
#pragma unroll
    for (int v = 0; v < 8; ++v) { rm[v] = -1e30f; rs[v] = 0.0f; }

    for (int j = 0; j < 64; ++j) {
      const __bf16* brow = Xb + (j * 16 + l16) * XB_STRIDE;
      const v16bf b0 = cat16(ld8(brow + hi * 8),      ld8(brow + 16 + hi * 8));
      const v16bf b1 = cat16(ld8(brow + 32 + hi * 8), ld8(brow + 48 + hi * 8));
      v8f c = {};
      c = WMMA_BF16(a0, b0, c);
      c = WMMA_BF16(a1, b1, c);
      const float uc = Us[j * 16 + l16];
#pragma unroll
      for (int v = 0; v < 8; ++v) {
        const float x  = c[v] + uc;
        const float nm = fmaxf(rm[v], x);
        rs[v] = rs[v] * __expf(rm[v] - nm) + __expf(x - nm);
        rm[v] = nm;
      }
    }
#pragma unroll
    for (int v = 0; v < 8; ++v) {
      float m = rm[v], s = rs[v];
#pragma unroll
      for (int mk = 1; mk < 16; mk <<= 1) {
        const float om = __shfl_xor(m, mk, 32);
        const float os = __shfl_xor(s, mk, 32);
        const float nm = fmaxf(m, om);
        s = s * __expf(m - nm) + os * __expf(om - nm);
        m = nm;
      }
      if (l16 == 0) Vs[R + v + 8 * hi] = logmu - (m + __logf(s));
    }
  }
  __syncthreads();

  // ======== Pass C: O = Q_ + (n * exp(S+u+v)) @ Q_ ========
  __bf16* psc = Psc + wave * (16 * P_STRIDE);
  for (int t = wave; t < 64; t += 16) {
    const int R = t * 16;
    const __bf16* arow = Xb + (R + l16) * XB_STRIDE;
    const v16bf a0 = cat16(ld8(arow + hi * 8),      ld8(arow + 16 + hi * 8));
    const v16bf a1 = cat16(ld8(arow + 32 + hi * 8), ld8(arow + 48 + hi * 8));
    float ur[8];
#pragma unroll
    for (int v = 0; v < 8; ++v) ur[v] = Us[R + v + 8 * hi];

    v8f acc[4] = {};
    for (int jp = 0; jp < 32; ++jp) {
#pragma unroll
      for (int sub = 0; sub < 2; ++sub) {
        const int j = jp * 2 + sub;
        const __bf16* brow = Xb + (j * 16 + l16) * XB_STRIDE;
        const v16bf b0 = cat16(ld8(brow + hi * 8),      ld8(brow + 16 + hi * 8));
        const v16bf b1 = cat16(ld8(brow + 32 + hi * 8), ld8(brow + 48 + hi * 8));
        v8f c = {};
        c = WMMA_BF16(a0, b0, c);
        c = WMMA_BF16(a1, b1, c);
        const float vc = Vs[j * 16 + l16];
#pragma unroll
        for (int v = 0; v < 8; ++v) {
          const float p = __expf(c[v] + ur[v] + vc + logn);  // pi * n
          psc[(v + 8 * hi) * P_STRIDE + sub * 16 + l16] = f2bf(p);
        }
      }
      // wave-private LDS relayout: D-layout stores -> A-layout loads
      asm volatile("s_wait_dscnt 0" ::: "memory");
      const v16bf pa = cat16(ld8(psc + l16 * P_STRIDE + hi * 8),
                             ld8(psc + l16 * P_STRIDE + 16 + hi * 8));
      const int jb = jp * 32;
#pragma unroll
      for (int cc = 0; cc < 4; ++cc) {
        // B[k][n] = X[jb+k][cc*16+n] = Xt[cc*16+n][jb+k]
        const __bf16* tb = Xt + (cc * 16 + l16) * XT_STRIDE + jb + hi * 8;
        const v16bf bb = cat16(ld8(tb), ld8(tb + 16));
        acc[cc] = WMMA_BF16(pa, bb, acc[cc]);
      }
    }

#pragma unroll
    for (int cc = 0; cc < 4; ++cc) {
#pragma unroll
      for (int v = 0; v < 8; ++v) {
        const int row = R + v + 8 * hi;
        const size_t idx = ((size_t)(b * 1024 + row)) * 512 + h * 64 + cc * 16 + l16;
        Opre[idx] = Qp[idx] + sc * acc[cc][v];   // undo 512^-0.25 on V side
      }
    }
  }
}

// ---------------------------------------------------------------------------
// LayerNorm over last dim (512). One block per row, 128 thr (4 elems each).
// ---------------------------------------------------------------------------
__global__ __launch_bounds__(128) void layernorm_kernel(
    const float* __restrict__ X, const float* __restrict__ g,
    const float* __restrict__ be, float* __restrict__ Y, __bf16* __restrict__ Ybf) {
  const int row = blockIdx.x;
  const int t   = threadIdx.x;
  const float4 vq = ((const float4*)(X + (size_t)row * 512))[t];
  float s  = vq.x + vq.y + vq.z + vq.w;
  float s2 = vq.x * vq.x + vq.y * vq.y + vq.z * vq.z + vq.w * vq.w;
#pragma unroll
  for (int m = 16; m >= 1; m >>= 1) {
    s  += __shfl_xor(s, m, 32);
    s2 += __shfl_xor(s2, m, 32);
  }
  __shared__ float sb[4], s2b[4];
  if ((t & 31) == 0) { sb[t >> 5] = s; s2b[t >> 5] = s2; }
  __syncthreads();
  s  = sb[0] + sb[1] + sb[2] + sb[3];
  s2 = s2b[0] + s2b[1] + s2b[2] + s2b[3];
  const float mean = s * (1.0f / 512.0f);
  const float var  = s2 * (1.0f / 512.0f) - mean * mean;
  const float rstd = rsqrtf(var + 1e-5f);
  const float4 gq = ((const float4*)g)[t];
  const float4 bq = ((const float4*)be)[t];
  float4 o;
  o.x = (vq.x - mean) * rstd * gq.x + bq.x;
  o.y = (vq.y - mean) * rstd * gq.y + bq.y;
  o.z = (vq.z - mean) * rstd * gq.z + bq.z;
  o.w = (vq.w - mean) * rstd * gq.w + bq.w;
  ((float4*)(Y + (size_t)row * 512))[t] = o;
  if (Ybf) {
    __bf16* yb = Ybf + (size_t)row * 512 + t * 4;
    yb[0] = f2bf(o.x); yb[1] = f2bf(o.y); yb[2] = f2bf(o.z); yb[3] = f2bf(o.w);
  }
}

// ---------------------------------------------------------------------------
// Launch pipeline
// ---------------------------------------------------------------------------
extern "C" void kernel_launch(void* const* d_in, const int* in_sizes, int n_in,
                              void* d_out, int out_size, void* d_ws, size_t ws_size,
                              hipStream_t stream) {
  (void)in_sizes; (void)n_in; (void)out_size; (void)ws_size;
  const float* Q  = (const float*)d_in[0];
  // d_in[1] = K (unused by reference)
  const float* Wq = (const float*)d_in[2];
  const float* bq = (const float*)d_in[3];
  const float* Wo = (const float*)d_in[4];
  const float* bo = (const float*)d_in[5];
  const float* g0 = (const float*)d_in[6];
  const float* b0 = (const float*)d_in[7];
  const float* g1 = (const float*)d_in[8];
  const float* b1 = (const float*)d_in[9];
  float* out = (float*)d_out;

  const int M = 8 * 1024;            // 8192 rows
  char* ws = (char*)d_ws;
  __bf16* Qbf  = (__bf16*)(ws + 0);                 //  8 MB
  __bf16* Wqbf = (__bf16*)(ws + 8388608);           //  0.5 MB
  __bf16* Wobf = (__bf16*)(ws + 8912896);           //  0.5 MB
  float*  Qp   = (float*)(ws + 9437184);            // 16 MB
  float*  Opre = (float*)(ws + 26214400);           // 16 MB
  float*  Xf   = (float*)(ws + 42991616);           // 16 MB (LN0 out, fp32)
  __bf16* Xbf  = (__bf16*)(ws + 59768832);          //  8 MB (LN0 out, bf16)
  float*  Yf   = (float*)(ws + 68157440);           // 16 MB
  __bf16* Xg   = (__bf16*)(ws + 84934656);          //  8 MB (head-major bf16 X)

  // 1) convert inputs to bf16
  cvt_bf16_kernel<<<dim3((M * 512 + 255) / 256), dim3(256), 0, stream>>>(Q, Qbf, M * 512);
  cvt_bf16_kernel<<<dim3((512 * 512 + 255) / 256), dim3(256), 0, stream>>>(Wq, Wqbf, 512 * 512);
  cvt_bf16_kernel<<<dim3((512 * 512 + 255) / 256), dim3(256), 0, stream>>>(Wo, Wobf, 512 * 512);

  // 2) Qp = Q @ Wq^T + bq
  gemm_bf16_kernel<<<dim3(M / 64, 4), dim3(256), 0, stream>>>(Qbf, Wqbf, bq, nullptr, Qp, 0);

  // 3) repack to head-major scaled bf16 (TDM source)
  repack_head_kernel<<<dim3(64), dim3(256), 0, stream>>>(Qp, Xg);

  // 4) per-(head,batch) Sinkhorn attention; X TDM-loaded into LDS
  attn_sinkhorn_kernel<<<dim3(64), dim3(512), ATTN_SMEM, stream>>>(Qp, Xg, Opre);

  // 5) LN0 (writes fp32 + bf16)
  layernorm_kernel<<<dim3(M), dim3(128), 0, stream>>>(Opre, g0, b0, Xf, Xbf);

  // 6) Y = X + relu(X @ Wo^T + bo)
  gemm_bf16_kernel<<<dim3(M / 64, 4), dim3(256), 0, stream>>>(Xbf, Wobf, bo, Xf, Yf, 1);

  // 7) LN1 -> output
  layernorm_kernel<<<dim3(M), dim3(128), 0, stream>>>(Yf, g1, b1, out, nullptr);
}
